// SeqAttention_1881195676194
// MI455X (gfx1250) — matrix operand order repaired and checked by reading, not compiled, not run on hardware
//
#include <hip/hip_runtime.h>

typedef __attribute__((ext_vector_type(16))) _Float16 v16h;
typedef __attribute__((ext_vector_type(8)))  _Float16 v8h;
typedef __attribute__((ext_vector_type(4)))  _Float16 v4h;
typedef __attribute__((ext_vector_type(8)))  float    v8f;
typedef __attribute__((ext_vector_type(4)))  int      v4i;

#define HIDDEN 128
#define SPAN   1024
#define BATCH  32
#define MQ     1024
#define KLEN   (MQ + SPAN)
#define NKV    (BATCH * KLEN * HIDDEN)   // elements in K or V
#define NTILE  (KLEN / 16)               // 128 key-tiles of 16

#if defined(__has_builtin)
#if __has_builtin(__builtin_amdgcn_global_load_async_to_lds_b128)
#define HAVE_ASYNC 1
#endif
#endif

typedef __attribute__((address_space(1))) v4i gv4i;   // global int4
typedef __attribute__((address_space(3))) v4i lv4i;   // LDS int4

static __device__ __forceinline__ gv4i* to_g(const void* p) {
    return (gv4i*)(uintptr_t)p;                       // generic global ptr == AS1 numerically
}
static __device__ __forceinline__ lv4i* to_l(void* p) {
    return (lv4i*)(uint32_t)(uintptr_t)p;             // generic LDS ptr low 32 bits == LDS offset
}

static __device__ __forceinline__ v8f wmma_f16(v16h a, v16h b, v8f c) {
    return __builtin_amdgcn_wmma_f32_16x16x32_f16(false, a, false, b, (short)0, c, false, false);
}

// ---------------- pre-pass: f32 -> f16 convert (K) ----------------
__global__ void cvt_f16_kernel(const float* __restrict__ in, _Float16* __restrict__ out, int n4) {
    const float4* in4 = (const float4*)in;
    v4h* out4 = (v4h*)out;
    for (int i = blockIdx.x * blockDim.x + threadIdx.x; i < n4; i += gridDim.x * blockDim.x) {
        float4 v = in4[i];
        v4h h; h[0] = (_Float16)v.x; h[1] = (_Float16)v.y; h[2] = (_Float16)v.z; h[3] = (_Float16)v.w;
        out4[i] = h;
    }
}

// ---------------- pre-pass: V -> f16, transposed per 16-key tile ----------------
// VhT[((b*NTILE + t)*HIDDEN + h)*16 + k] = V[b][t*16+k][h]
__global__ __launch_bounds__(256) void vtrans_kernel(const float* __restrict__ V, _Float16* __restrict__ VhT) {
    __shared__ _Float16 tile[16 * HIDDEN];
    const int bt = blockIdx.x;             // b*NTILE + t
    const int tid = threadIdx.x;
    const float* src = V + (size_t)bt * 16 * HIDDEN;   // 16 rows of HIDDEN
    {   // load 2048 floats, 8 per thread, coalesced
        int li = tid * 8;
        const float4* s4 = (const float4*)(src + li);
        float4 a = s4[0], b = s4[1];
        tile[li+0] = (_Float16)a.x; tile[li+1] = (_Float16)a.y;
        tile[li+2] = (_Float16)a.z; tile[li+3] = (_Float16)a.w;
        tile[li+4] = (_Float16)b.x; tile[li+5] = (_Float16)b.y;
        tile[li+6] = (_Float16)b.z; tile[li+7] = (_Float16)b.w;
    }
    __syncthreads();
    {   // write [h][k] layout, 8 consecutive k per thread
        int lo = tid * 8;
        int h  = lo >> 4;
        int k0 = lo & 15;
        v8h o;
        #pragma unroll
        for (int j = 0; j < 8; ++j) o[j] = tile[(k0 + j) * HIDDEN + h];
        *(v8h*)(VhT + ((size_t)bt * HIDDEN + h) * 16 + k0) = o;
    }
}

// ---------------- pre-pass: P[h][l] -> Pt[l][h] (f16) ----------------
__global__ void ptrans_kernel(const float* __restrict__ P, _Float16* __restrict__ Pt) {
    int idx = blockIdx.x * blockDim.x + threadIdx.x;   // over SPAN*HIDDEN
    if (idx < SPAN * HIDDEN) {
        int l = idx >> 7, h = idx & 127;
        Pt[idx] = (_Float16)P[(size_t)h * SPAN + l];
    }
}

// ---------------- main: sliding-window rel-pos flash attention ----------------
__global__ __launch_bounds__(32)
void relattn_kernel(const float*    __restrict__ Q,
                    const _Float16* __restrict__ Kh,   // [B][KLEN][H] f16
                    const _Float16* __restrict__ VhT,  // [B][NTILE][H][16] f16
                    const _Float16* __restrict__ Pt,   // [SPAN][H] f16
                    float* __restrict__ Out)
{
    __shared__ _Float16 posBuf[16 * SPAN];          // 32 KB
    __shared__ _Float16 vbuf[2 * 2 * HIDDEN * 16];  // 16 KB: [buf][kh][h][k16]
    __shared__ _Float16 pbuf[16 * 32];              //  1 KB

    const int lane = threadIdx.x;
    const int col  = lane & 15;
    const int half = lane >> 4;
    const int m0   = blockIdx.x * 16;
    const int b    = blockIdx.y;
    const float scale = 0.08838834764831845f;       // 1/sqrt(128)

    const float*    Qb  = Q   + (size_t)b * MQ * HIDDEN;
    const _Float16* Kb  = Kh  + (size_t)b * KLEN * HIDDEN;
    const _Float16* Vtb = VhT + (size_t)b * NTILE * HIDDEN * 16;

    // ---- Q tile -> WMMA A operands (A layout: M=lane&15; e->K = 16h+e)
    v16h aQ[4];
    {
        const float* qrow = Qb + (size_t)(m0 + col) * HIDDEN;
        #pragma unroll
        for (int c = 0; c < 4; ++c) {
            #pragma unroll
            for (int e = 0; e < 16; ++e) {
                int k = half * 16 + e;
                aQ[c][e] = (_Float16)qrow[c * 32 + k];
            }
        }
    }

    // ---- positional logits posBuf[i][l] = Q[m0+i] . P[:,l]
    for (int ct = 0; ct < SPAN / 16; ++ct) {
        const int l = ct * 16 + col;
        v8f acc = {};
        #pragma unroll
        for (int c = 0; c < 4; ++c) {
            // B layout: N=lane&15, e -> K = half*16+e; contiguous h in Pt[l][...]
            v16h bP = *(const v16h*)(Pt + (size_t)l * HIDDEN + c * 32 + half * 16);
            acc = wmma_f16(aQ[c], bP, acc);
        }
        #pragma unroll
        for (int r = 0; r < 8; ++r)
            posBuf[(r + 8 * half) * SPAN + ct * 16 + col] = (_Float16)acc[r];
    }
    __syncthreads();

    // ---- flash state
    float m8[8], s8[8];
    v8f   Oc[8];
    #pragma unroll
    for (int r = 0; r < 8; ++r) { m8[r] = -3.0e38f; s8[r] = 0.0f; }
    #pragma unroll
    for (int c = 0; c < 8; ++c) { Oc[c] = v8f{}; }

    const int NSTEP = 33;

    // ---- stage V tiles for one step into vbuf[buf] (2 tiles of 128x16 f16 = 4KB each)
    auto stageV = [&](int buf, int s) {
        const int t0 = (m0 + 32 * s) >> 4;
        #pragma unroll
        for (int kh = 0; kh < 2; ++kh) {
            int t = t0 + kh; if (t > NTILE - 1) t = NTILE - 1;
            const _Float16* g = Vtb + (size_t)t * (HIDDEN * 16) + lane * 8;
            _Float16* d = &vbuf[(buf * 2 + kh) * (HIDDEN * 16) + lane * 8];
#ifdef HAVE_ASYNC
            #pragma unroll
            for (int i = 0; i < 8; ++i)
                __builtin_amdgcn_global_load_async_to_lds_b128(to_g(g + i * 256), to_l(d + i * 256), 0, 0);
#else
            #pragma unroll
            for (int i = 0; i < 8; ++i)
                *(v8h*)(d + i * 256) = *(const v8h*)(g + i * 256);
#endif
        }
    };

    stageV(0, 0);

    for (int s = 0; s < NSTEP; ++s) {
        const int kbase = m0 + 32 * s;
        if (s + 1 < NSTEP) stageV((s + 1) & 1, s + 1);

        // ---- scores S[16x32] = Qtile @ K^T
        v8f Sa = {}, Sb = {};
        #pragma unroll
        for (int kh = 0; kh < 2; ++kh) {
            int kabs = kbase + kh * 16 + col;
            int kcl  = kabs > KLEN - 1 ? KLEN - 1 : kabs;
            const _Float16* krow = Kb + (size_t)kcl * HIDDEN;
            #pragma unroll
            for (int c = 0; c < 4; ++c) {
                v16h bK = *(const v16h*)(krow + c * 32 + half * 16);
                if (kh == 0) Sa = wmma_f16(aQ[c], bK, Sa);
                else         Sb = wmma_f16(aQ[c], bK, Sb);
            }
        }
        {   // prefetch next step's K rows
            int kn = kbase + 32 + col; if (kn > KLEN - 1) kn = KLEN - 1;
            __builtin_prefetch(Kb + (size_t)kn * HIDDEN, 0, 0);
        }

        // ---- bias + mask + online softmax
        #pragma unroll
        for (int r = 0; r < 8; ++r) {
            const int i  = r + 8 * half;
            const int la = 32 * s + col - i;
            const int lb = la + 16;
            int lac = la < 0 ? 0 : (la > SPAN - 1 ? SPAN - 1 : la);
            int lbc = lb < 0 ? 0 : (lb > SPAN - 1 ? SPAN - 1 : lb);
            bool va = (la >= 0) && (la < SPAN);
            bool vb = (lb >= 0) && (lb < SPAN);
            float xa = va ? (Sa[r] + (float)posBuf[i * SPAN + lac]) * scale : -3.0e38f;
            float xb = vb ? (Sb[r] + (float)posBuf[i * SPAN + lbc]) * scale : -3.0e38f;

            float t = fmaxf(xa, xb);
            #pragma unroll
            for (int off = 8; off >= 1; off >>= 1)
                t = fmaxf(t, __shfl_xor(t, off, 32));
            float nm = fmaxf(m8[r], t);
            float al = __expf(m8[r] - nm);
            float pa = __expf(xa - nm);
            float pb = __expf(xb - nm);
            float ps = pa + pb;
            #pragma unroll
            for (int off = 8; off >= 1; off >>= 1)
                ps += __shfl_xor(ps, off, 32);
            s8[r] = s8[r] * al + ps;
            m8[r] = nm;
            #pragma unroll
            for (int c = 0; c < 8; ++c) Oc[c][r] *= al;

            pbuf[i * 32 + col]      = (_Float16)pa;
            pbuf[i * 32 + 16 + col] = (_Float16)pb;
        }

#ifdef HAVE_ASYNC
        if (s + 1 < NSTEP) { asm volatile("s_wait_asynccnt 0x10" ::: "memory"); }
        else               { asm volatile("s_wait_asynccnt 0x0"  ::: "memory"); }
#endif

        // ---- O += P(16x32) @ V(32x128)
        v16h aP;
        {   // A layout: one contiguous 16-element run in pbuf[row][j], e -> K = half*16+e
            v8h lo = *(const v8h*)(pbuf + col * 32 + half * 16);
            v8h hi = *(const v8h*)(pbuf + col * 32 + half * 16 + 8);
            aP = __builtin_shufflevector(lo, hi, 0,1,2,3,4,5,6,7,8,9,10,11,12,13,14,15);
        }
        const _Float16* vb0 = &vbuf[(s & 1) * 2 * (HIDDEN * 16)];
        #pragma unroll
        for (int c = 0; c < 8; ++c) {
            // B layout: N = col (h), e -> K = half*16+e; vbuf[kh=half][h][k16] is contiguous in e
            v16h bV = *(const v16h*)(vb0 + (size_t)half * (HIDDEN * 16) + (c * 16 + col) * 16);
            Oc[c] = wmma_f16(aP, bV, Oc[c]);
        }
    }

    // ---- normalize & store
    float* orow = Out + ((size_t)b * MQ + m0) * HIDDEN;
    #pragma unroll
    for (int r = 0; r < 8; ++r) {
        const int i = r + 8 * half;
        const float inv = 1.0f / s8[r];
        #pragma unroll
        for (int c = 0; c < 8; ++c)
            orow[(size_t)i * HIDDEN + c * 16 + col] = Oc[c][r] * inv;
    }
}

extern "C" void kernel_launch(void* const* d_in, const int* in_sizes, int n_in,
                              void* d_out, int out_size, void* d_ws, size_t ws_size,
                              hipStream_t stream) {
    (void)in_sizes; (void)n_in; (void)out_size; (void)ws_size;
    const float* Q = (const float*)d_in[0];
    const float* K = (const float*)d_in[1];
    const float* V = (const float*)d_in[2];
    const float* P = (const float*)d_in[3];
    float*       O = (float*)d_out;

    _Float16* Kh  = (_Float16*)d_ws;           // NKV elements
    _Float16* VhT = Kh + (size_t)NKV;          // NKV elements
    _Float16* Pt  = VhT + (size_t)NKV;         // SPAN*HIDDEN elements

    cvt_f16_kernel<<<2048, 256, 0, stream>>>(K, Kh, NKV / 4);
    vtrans_kernel<<<BATCH * NTILE, 256, 0, stream>>>(V, VhT);
    ptrans_kernel<<<(SPAN * HIDDEN + 255) / 256, 256, 0, stream>>>(P, Pt);

    dim3 grid(MQ / 16, BATCH);
    relattn_kernel<<<grid, dim3(32), 0, stream>>>(Q, Kh, VhT, Pt, O);
}
